// Encoder_22273700397058
// MI455X (gfx1250) — matrix operand order repaired and checked by reading, not compiled, not run on hardware
//
#include <hip/hip_runtime.h>
#include <hip/hip_bf16.h>
#include <math.h>

#define NN     50000
#define GG     100
#define EE     500000
#define DIMV   128
#define RL2    100000
#define NPER   500
#define SA     136      // padded LDS row stride in bf16 elems (avoids bank conflicts)

typedef __attribute__((ext_vector_type(16))) __bf16 v16bf;
typedef __attribute__((ext_vector_type(8)))  float  v8f;

union AFu { v16bf v; unsigned u[8]; };

// A fragment (16x32 bf16, contiguous k order, same as B): lane l=(h,r): K(e,h) = e+16h, row=M.
// Pairs of consecutive K -> 32-bit LDS loads at bf16 offsets {0,2,4,6,8,10,12,14}.
__device__ __forceinline__ v16bf load_a_frag(const __bf16* lds, int row, int kb, int h) {
    AFu a;
    const unsigned* p32 = (const unsigned*)(lds + row * SA + kb * 32 + 16 * h);
    a.u[0] = p32[0]; a.u[1] = p32[1]; a.u[2] = p32[2]; a.u[3] = p32[3];
    a.u[4] = p32[4]; a.u[5] = p32[5]; a.u[6] = p32[6]; a.u[7] = p32[7];
    return a.v;
}

// ---------------- weight prep: f32 [k][n] -> bf16 transposed [n][k] -------------
__global__ void prep_k(const float* Wi1, const float* Wi2, const float* Wg1,
                       const float* Wg2, __bf16* Wt) {
    int m = blockIdx.x;
    const float* src = (m == 0) ? Wi1 : (m == 1) ? Wi2
                     : (m < 5) ? (Wg1 + (m - 2) * 16384) : (Wg2 + (m - 5) * 16384);
    __bf16* dst = Wt + m * 16384;
    for (int i = threadIdx.x; i < 16384; i += 256) {
        int n = i >> 7, k = i & 127;
        dst[i] = (__bf16)src[k * DIMV + n];
    }
}

// ---------------- fused MLP: OUT = [relu]( relu((X[+AGG])@W1+b1) @ W2 + b2 ) ----
// One wave per 16-row tile, 8 waves/block -> 128 rows/block. bf16 WMMA, f32 acc.
__global__ __launch_bounds__(256) void mlp_wmma(
    const float* __restrict__ X, const float* AGG,
    const __bf16* __restrict__ W1t, const float* __restrict__ B1,
    const __bf16* __restrict__ W2t, const float* __restrict__ B2,
    float* OUT, int R, int relu_out, float* gsum, float* gsq) {
    __shared__ __bf16 lA[128 * SA];
    __shared__ float sSum[DIMV];
    __shared__ float sSq[DIMV];
    const int t = threadIdx.x;
    const long row0 = (long)blockIdx.x * 128;

    if (gsum != nullptr && t < DIMV) { sSum[t] = 0.f; sSq[t] = 0.f; }

    { // stage (X [+ AGG]) tile into LDS as bf16; each thread: 1/2 row, coalesced
        int r = t >> 1;
        int c0 = (t & 1) * 64;
        long grow = row0 + r;
        __bf16* dst = lA + r * SA + c0;
        if (grow < R) {
            const float4* xp = (const float4*)(X + grow * DIMV + c0);
            const float4* ap = AGG ? (const float4*)(AGG + grow * DIMV + c0) : nullptr;
#pragma unroll
            for (int i = 0; i < 16; i++) {
                float4 v = xp[i];
                if (ap) { float4 a = ap[i]; v.x += a.x; v.y += a.y; v.z += a.z; v.w += a.w; }
                dst[i * 4 + 0] = (__bf16)v.x; dst[i * 4 + 1] = (__bf16)v.y;
                dst[i * 4 + 2] = (__bf16)v.z; dst[i * 4 + 3] = (__bf16)v.w;
            }
        } else {
#pragma unroll
            for (int i = 0; i < 64; i++) dst[i] = (__bf16)0.f;
        }
    }
    __syncthreads();

    const int w = t >> 5, l = t & 31, h = l >> 4, r = l & 15;
    const int tile0 = w * 16;
    v8f acc[8];

    // GEMM 1: C init = bias broadcast (C layout: N = lane%16 for every VGPR)
#pragma unroll
    for (int nb = 0; nb < 8; nb++) {
        float b = B1[nb * 16 + r];
#pragma unroll
        for (int j = 0; j < 8; j++) acc[nb][j] = b;
    }
#pragma unroll
    for (int kb = 0; kb < 4; kb++) {
        v16bf af = load_a_frag(lA, tile0 + r, kb, h);
#pragma unroll
        for (int nb = 0; nb < 8; nb++) {
            v16bf bf = *(const v16bf*)(W1t + (nb * 16 + r) * DIMV + kb * 32 + h * 16);
            acc[nb] = __builtin_amdgcn_wmma_f32_16x16x32_bf16(
                false, af, false, bf, (short)0, acc[nb], false, false);
        }
    }
    // ReLU + write intermediate tile back in place (wave touches only its own rows)
#pragma unroll
    for (int nb = 0; nb < 8; nb++) {
#pragma unroll
        for (int j = 0; j < 8; j++) {
            float v = acc[nb][j];
            v = v > 0.f ? v : 0.f;
            lA[(tile0 + j + 8 * h) * SA + nb * 16 + r] = (__bf16)v;
        }
    }
    __syncthreads();

    // GEMM 2
#pragma unroll
    for (int nb = 0; nb < 8; nb++) {
        float b = B2[nb * 16 + r];
#pragma unroll
        for (int j = 0; j < 8; j++) acc[nb][j] = b;
    }
#pragma unroll
    for (int kb = 0; kb < 4; kb++) {
        v16bf af = load_a_frag(lA, tile0 + r, kb, h);
#pragma unroll
        for (int nb = 0; nb < 8; nb++) {
            v16bf bf = *(const v16bf*)(W2t + (nb * 16 + r) * DIMV + kb * 32 + h * 16);
            acc[nb] = __builtin_amdgcn_wmma_f32_16x16x32_bf16(
                false, af, false, bf, (short)0, acc[nb], false, false);
        }
    }

    // write out (+ optional relu) and accumulate per-dim BN partial sums
#pragma unroll
    for (int nb = 0; nb < 8; nb++) {
        int col = nb * 16 + r;
        float s = 0.f, q = 0.f;
#pragma unroll
        for (int j = 0; j < 8; j++) {
            long grow = row0 + tile0 + j + 8 * h;
            float v = acc[nb][j];
            if (relu_out) v = v > 0.f ? v : 0.f;
            if (grow < R) {
                OUT[grow * DIMV + col] = v;
                s += v; q += v * v;
            }
        }
        if (gsum) { atomicAdd(&sSum[col], s); atomicAdd(&sSq[col], q); }
    }
    __syncthreads();
    if (gsum && t < DIMV) { atomicAdd(&gsum[t], sSum[t]); atomicAdd(&gsq[t], sSq[t]); }
}

// ---------------- elementwise helpers ------------------------------------------
__global__ void zero_k(float* p, long n) {
    for (long i = (long)blockIdx.x * blockDim.x + threadIdx.x; i < n;
         i += (long)gridDim.x * blockDim.x) p[i] = 0.f;
}
__global__ void fill_k(float* p, long n, const float* v) {
    float x = *v;
    for (long i = (long)blockIdx.x * blockDim.x + threadIdx.x; i < n;
         i += (long)gridDim.x * blockDim.x) p[i] = x;
}

// h[N + i] = h[neg_idx[i]]
__global__ void gather_k(float* h, const int* __restrict__ neg) {
    long tid = (long)blockIdx.x * 256 + threadIdx.x;   // NN*32 threads exactly
    long node = tid >> 5; int c = (int)(tid & 31) * 4;
    long s = neg[node];
    *(float4*)(h + ((long)NN + node) * DIMV + c) = *(const float4*)(h + s * DIMV + c);
}

// agg[dst2[e]] += h[src2[e]]   (doubled edge list, offset N for second copy)
__global__ void scatter_k(const float* __restrict__ h, float* agg,
                          const int* __restrict__ src, const int* __restrict__ dst) {
    long tid = (long)blockIdx.x * 256 + threadIdx.x;   // 2E*32 threads exactly
    long e = tid >> 5; int lane = (int)(tid & 31);
    int s, d;
    if (e < EE) { s = src[e]; d = dst[e]; }
    else        { s = src[e - EE] + NN; d = dst[e - EE] + NN; }
    float4 v = *(const float4*)(h + (long)s * DIMV + lane * 4);
    float* ap = agg + (long)d * DIMV + lane * 4;
    atomicAdd(ap + 0, v.x); atomicAdd(ap + 1, v.y);
    atomicAdd(ap + 2, v.z); atomicAdd(ap + 3, v.w);
}

__global__ void bnfin_k(const float* gsum, const float* gsq, const float* gamma,
                        const float* beta, float* scale, float* shift) {
    int t = threadIdx.x;
    if (t < DIMV) {
        float mu  = gsum[t] / (float)RL2;
        float var = gsq[t] / (float)RL2 - mu * mu;
        float sc  = gamma[t] * rsqrtf(var + 1e-5f);
        scale[t] = sc;
        shift[t] = beta[t] - mu * sc;
    }
}

// h = z*scale+shift ; glob += w_layer[i]*h
__global__ void bnapply_k(const float* __restrict__ z, float* __restrict__ h,
                          float* __restrict__ glob, const float* __restrict__ scale,
                          const float* __restrict__ shift, const float* wl) {
    long tid = (long)blockIdx.x * 256 + threadIdx.x;   // RL2*32 threads exactly
    long row = tid >> 5; int c = (int)(tid & 31) * 4;
    float w = *wl;
    float4 zv = *(const float4*)(z + row * DIMV + c);
    float4 sc = *(const float4*)(scale + c);
    float4 sh = *(const float4*)(shift + c);
    float4 hv; hv.x = zv.x * sc.x + sh.x; hv.y = zv.y * sc.y + sh.y;
               hv.z = zv.z * sc.z + sh.z; hv.w = zv.w * sc.w + sh.w;
    *(float4*)(h + row * DIMV + c) = hv;
    float4 gv = *(float4*)(glob + row * DIMV + c);
    gv.x += w * hv.x; gv.y += w * hv.y; gv.z += w * hv.z; gv.w += w * hv.w;
    *(float4*)(glob + row * DIMV + c) = gv;
}

// pos_graph / neg_graph segment sums (contiguous 500-node graphs)
__global__ void seg_k(const float* __restrict__ glob, float* __restrict__ outp,
                      float* __restrict__ outn) {
    int b = blockIdx.x, d = threadIdx.x;
    int g = b < GG ? b : b - GG;
    const float* base = glob + ((b < GG ? 0L : (long)NN) + (long)g * NPER) * DIMV + d;
    float s = 0.f;
    for (int i = 0; i < NPER; i++) s += base[(long)i * DIMV];
    (b < GG ? outp : outn)[g * DIMV + d] = s;
}

// pooling tree: softmax(2) -> tanh(|y0-y1|/2) and sign test; node routes to 1 of 4 leaves
__global__ __launch_bounds__(256) void pool_k(
    const float* __restrict__ glob, const float* __restrict__ Wp,
    const float* __restrict__ bp, const float* __restrict__ wsub,
    float* samp, float* probacc) {
    int w = (blockIdx.x << 3) + (threadIdx.x >> 5);
    int lane = threadIdx.x & 31;
    if (w >= NN) return;
    const float* row = glob + (long)w * DIMV;
    float v0 = row[lane], v1 = row[lane + 32], v2 = row[lane + 64], v3 = row[lane + 96];

    bool m0, m1, m2;
    float diffs = 0.f;
    auto proj = [&](float mask, int p, bool& m) -> float {
        const float* Wb = Wp + p * 256;
        float y0 = v0 * mask * Wb[lane * 2]       + v1 * mask * Wb[(lane + 32) * 2]
                 + v2 * mask * Wb[(lane + 64) * 2] + v3 * mask * Wb[(lane + 96) * 2];
        float y1 = v0 * mask * Wb[lane * 2 + 1]       + v1 * mask * Wb[(lane + 32) * 2 + 1]
                 + v2 * mask * Wb[(lane + 64) * 2 + 1] + v3 * mask * Wb[(lane + 96) * 2 + 1];
#pragma unroll
        for (int off = 16; off > 0; off >>= 1) {
            y0 += __shfl_xor(y0, off, 32);
            y1 += __shfl_xor(y1, off, 32);
        }
        y0 += bp[p * 2]; y1 += bp[p * 2 + 1];
        m = (y0 >= y1);
        return tanhf(0.5f * fabsf(y0 - y1));
    };
    diffs += proj(1.f, 0, m0);
    diffs += proj(m0 ? 1.f : 0.f, 1, m1);
    diffs += proj(m0 ? 0.f : 1.f, 2, m2);
    int s = m0 ? (m1 ? 0 : 1) : (m2 ? 2 : 3);
    float wsv = wsub[s];
    int g = w / NPER;
    float* sp = samp + g * DIMV;
    atomicAdd(sp + lane,      wsv * v0);
    atomicAdd(sp + lane + 32, wsv * v1);
    atomicAdd(sp + lane + 64, wsv * v2);
    atomicAdd(sp + lane + 96, wsv * v3);
    if (lane == 0) atomicAdd(probacc, diffs);
}

// InfoNCE: counts are all 500 -> info_nce = mean over nodes of per-node loss
__global__ __launch_bounds__(256) void infonce_k(const float* __restrict__ glob,
                                                 const float* __restrict__ posg,
                                                 float* nceacc) {
    __shared__ float nk[GG * 132];     // padded stride vs bank conflicts
    __shared__ float rnorm[GG];
    __shared__ float qbuf[8 * DIMV];
    __shared__ float blockloss;
    int t = threadIdx.x;
    if (t == 0) blockloss = 0.f;
    if (t < GG) {
        float s = 0.f;
        for (int d = 0; d < DIMV; d++) { float x = posg[t * DIMV + d]; s += x * x; }
        rnorm[t] = 1.f / fmaxf(sqrtf(s), 1e-12f);
    }
    __syncthreads();
    for (int idx = t; idx < GG * DIMV; idx += 256) {
        int g = idx >> 7, d = idx & 127;
        nk[g * 132 + d] = posg[idx] * rnorm[g];
    }
    __syncthreads();

    int w = t >> 5, lane = t & 31;
    float* qw = qbuf + w * DIMV;
    bool has3 = lane < 4;
    for (long node = (long)blockIdx.x * 8 + w; node < NN; node += (long)gridDim.x * 8) {
        const float* row = glob + node * DIMV;
        float v0 = row[lane], v1 = row[lane + 32], v2 = row[lane + 64], v3 = row[lane + 96];
        float s = v0 * v0 + v1 * v1 + v2 * v2 + v3 * v3;
#pragma unroll
        for (int off = 16; off > 0; off >>= 1) s += __shfl_xor(s, off, 32);
        float isc = 1.f / fmaxf(sqrtf(s), 1e-12f);
        qw[lane] = v0 * isc; qw[lane + 32] = v1 * isc;
        qw[lane + 64] = v2 * isc; qw[lane + 96] = v3 * isc;
        __builtin_amdgcn_wave_barrier();
        float t0a = 0.f, t1a = 0.f, t2a = 0.f, t3a = 0.f;
        for (int d = 0; d < DIMV; d++) {
            float q = qw[d];
            t0a += q * nk[lane * 132 + d];
            t1a += q * nk[(lane + 32) * 132 + d];
            t2a += q * nk[(lane + 64) * 132 + d];
            if (has3) t3a += q * nk[(lane + 96) * 132 + d];
        }
        t0a *= 10.f; t1a *= 10.f; t2a *= 10.f; t3a *= 10.f;   // 1/TEMP
        float mx = fmaxf(fmaxf(t0a, t1a), t2a);
        if (has3) mx = fmaxf(mx, t3a);
#pragma unroll
        for (int off = 16; off > 0; off >>= 1) mx = fmaxf(mx, __shfl_xor(mx, off, 32));
        float se = expf(t0a - mx) + expf(t1a - mx) + expf(t2a - mx);
        if (has3) se += expf(t3a - mx);
#pragma unroll
        for (int off = 16; off > 0; off >>= 1) se += __shfl_xor(se, off, 32);
        float tz = __shfl(t0a, 0, 32);          // dot with graph 0 (duplicated col)
        float loss = mx + logf(se + expf(tz - mx)) - tz;
        if (lane == 0) atomicAdd(&blockloss, loss);
        __builtin_amdgcn_wave_barrier();
    }
    __syncthreads();
    if (t == 0) atomicAdd(nceacc, blockloss);
}

__global__ void finalize_k(const float* probacc, const float* nceacc, float* out) {
    if (threadIdx.x == 0) {
        out[38400] = -probacc[0];
        out[38401] = nceacc[0] / (float)NN;
    }
}

extern "C" void kernel_launch(void* const* d_in, const int* in_sizes, int n_in,
                              void* d_out, int out_size, void* d_ws, size_t ws_size,
                              hipStream_t stream) {
    const float* x        = (const float*)d_in[0];
    const int*   edge     = (const int*)d_in[1];
    const int*   neg      = (const int*)d_in[3];
    const float* W_ini1   = (const float*)d_in[5];
    const float* b_ini1   = (const float*)d_in[6];
    const float* W_ini2   = (const float*)d_in[7];
    const float* b_ini2   = (const float*)d_in[8];
    const float* W_gin1   = (const float*)d_in[9];
    const float* b_gin1   = (const float*)d_in[10];
    const float* W_gin2   = (const float*)d_in[11];
    const float* b_gin2   = (const float*)d_in[12];
    const float* bn_gamma = (const float*)d_in[13];
    const float* bn_beta  = (const float*)d_in[14];
    const float* w_layer  = (const float*)d_in[15];
    const float* b_layer  = (const float*)d_in[16];
    const float* Wp       = (const float*)d_in[17];
    const float* bp       = (const float*)d_in[18];
    const float* w_sub    = (const float*)d_in[19];
    const float* b_sub    = (const float*)d_in[20];
    float* out = (float*)d_out;

    float*  ws      = (float*)d_ws;
    float*  hbuf    = ws;                       // 2N*128
    float*  aggbuf  = ws + 12800000L;           // 2N*128 (also holds z in place)
    float*  globbuf = ws + 25600000L;           // 2N*128
    __bf16* wtbuf   = (__bf16*)(ws + 38400000L);// 8 * 128*128 bf16 transposed weights
    float*  gsum    = (float*)(wtbuf + 8 * 16384);
    float*  gsq     = gsum + 128;
    float*  bscale  = gsq + 128;
    float*  bshift  = bscale + 128;
    float*  probacc = bshift + 128;
    float*  nceacc  = probacc + 1;

    const int* srcv = edge;
    const int* dstv = edge + EE;

    prep_k<<<8, 256, 0, stream>>>(W_ini1, W_ini2, W_gin1, W_gin2, wtbuf);

    // h0 = relu(x@W1+b1)@W2+b2 ; then h[N:] = h0[neg_idx]
    mlp_wmma<<<(NN + 127) / 128, 256, 0, stream>>>(
        x, nullptr, wtbuf, b_ini1, wtbuf + 16384, b_ini2, hbuf, NN, 0, nullptr, nullptr);
    gather_k<<<NN * 32 / 256, 256, 0, stream>>>(hbuf, neg);

    fill_k<<<2048, 256, 0, stream>>>(globbuf, 12800000L, b_layer);  // glob = b_layer

    for (int i = 0; i < 3; i++) {
        zero_k<<<2048, 256, 0, stream>>>(aggbuf, 12800000L);
        zero_k<<<1, 256, 0, stream>>>(gsum, 256L);                  // gsum + gsq
        scatter_k<<<(2 * EE * 32) / 256, 256, 0, stream>>>(hbuf, aggbuf, srcv, dstv);
        mlp_wmma<<<(RL2 + 127) / 128, 256, 0, stream>>>(
            hbuf, aggbuf,
            wtbuf + (2 + i) * 16384, b_gin1 + i * 128,
            wtbuf + (5 + i) * 16384, b_gin2 + i * 128,
            aggbuf, RL2, 1, gsum, gsq);                             // z in place of agg
        bnfin_k<<<1, 128, 0, stream>>>(gsum, gsq, bn_gamma + i * 128,
                                       bn_beta + i * 128, bscale, bshift);
        bnapply_k<<<RL2 * 32 / 256, 256, 0, stream>>>(
            aggbuf, hbuf, globbuf, bscale, bshift, w_layer + i);
    }

    // outputs: [pos_graph | sampling_subgraph | neg_graph | prob_loss | info_nce]
    seg_k<<<200, 128, 0, stream>>>(globbuf, out, out + 25600);
    fill_k<<<64, 256, 0, stream>>>(out + 12800, 12800L, b_sub);     // samp = b_sub
    zero_k<<<1, 32, 0, stream>>>(probacc, 2L);
    pool_k<<<(NN + 7) / 8, 256, 0, stream>>>(globbuf, Wp, bp, w_sub, out + 12800, probacc);
    infonce_k<<<128, 256, 0, stream>>>(globbuf, out, nceacc);
    finalize_k<<<1, 32, 0, stream>>>(probacc, nceacc, out);
}
